// WSDDN_res_65798898975310
// MI455X (gfx1250) — matrix operand order repaired and checked by reading, not compiled, not run on hardware
//
#include <hip/hip_runtime.h>
#include <hip/hip_bf16.h>
#include <math.h>

// ---------------------------------------------------------------------------
// WSDDN (ResNet-34 backbone + ROI-SPP + FC stack) for MI455X / gfx1250.
// All matrix work (convs as implicit GEMM, FC layers) runs through
// V_WMMA_F32_16X16X32_BF16 (bf16 operands, f32 accumulate). K is staged in
// 64-wide chunks through LDS (2 WMMAs per barrier), fragments are loaded as
// 16-byte LDS vectors matching the documented wave32 WMMA register layouts.
// Gathers use clamp-and-select (never predicated loads) so the memory
// pipeline can batch requests under a single wait.
// ---------------------------------------------------------------------------

typedef __bf16 bf16_t;
typedef __attribute__((ext_vector_type(16))) __bf16 v16bf;
typedef __attribute__((ext_vector_type(8)))  __bf16 v8bf;
typedef __attribute__((ext_vector_type(8)))  float  v8f;

#define TM 64            // Cout rows per workgroup (16 per wave, 4 waves)
#define TN 16            // output pixels / GEMM rows per workgroup
#define KC 64            // K-chunk staged per barrier (2 WMMAs of K=32)
#define RS (KC + 8)      // padded LDS row (bf16 elems); 144 B keeps 16B align

__device__ __forceinline__ int iclamp(int x, int lo, int hi) {
  return x < lo ? lo : (x > hi ? hi : x);
}

// ---------------------------------------------------------------------------
// fp32 -> bf16 conversion
// ---------------------------------------------------------------------------
__global__ void cvt_f32_bf16(const float* __restrict__ in,
                             bf16_t* __restrict__ out, int n) {
  int i = blockIdx.x * blockDim.x + threadIdx.x;
  if (i < n) out[i] = (bf16_t)in[i];
}

// ---------------------------------------------------------------------------
// Shared device helpers
// ---------------------------------------------------------------------------
__device__ __forceinline__ v8bf pack8(float4 f0, float4 f1) {
  v8bf t;
  t[0] = (bf16_t)f0.x; t[1] = (bf16_t)f0.y; t[2] = (bf16_t)f0.z; t[3] = (bf16_t)f0.w;
  t[4] = (bf16_t)f1.x; t[5] = (bf16_t)f1.y; t[6] = (bf16_t)f1.z; t[7] = (bf16_t)f1.w;
  return t;
}

#define MERGE16(lo, hi) \
  __builtin_shufflevector(lo, hi, 0, 1, 2, 3, 4, 5, 6, 7, 8, 9, 10, 11, 12, 13, 14, 15)

// ---------------------------------------------------------------------------
// Fused conv + eval-BN (+residual) (+ReLU), implicit GEMM with WMMA bf16.
// A = W[Cout][Cin*KH*KW] (OIHW is K-contiguous), B = im2col gather (n-major
// in LDS so fragments are contiguous).
// ---------------------------------------------------------------------------
__global__ __launch_bounds__(128) void conv_bn_wmma(
    const bf16_t* __restrict__ in, bf16_t* __restrict__ out,
    const float* __restrict__ w,
    const float* __restrict__ bng, const float* __restrict__ bnb,
    const float* __restrict__ bnm, const float* __restrict__ bnv,
    const bf16_t* __restrict__ res, int relu,
    int Cin, int Cout, int Hin, int Win, int Hout, int Wout,
    int KH, int KW, int stride, int pad) {
  __shared__ bf16_t As[TM][RS];    // [m][k]
  __shared__ bf16_t Bs[TN][RS];    // [n][k]  (n-major for contiguous frags)

  const int tid  = threadIdx.x;
  const int wave = tid >> 5;
  const int lane = tid & 31;
  const int n0 = blockIdx.x * TN;
  const int m0 = blockIdx.y * TM;
  const int K  = Cin * KH * KW;
  const int HW = Hout * Wout;
  const int khw = KH * KW;
  const bool krow_aligned = ((K & 3) == 0);  // float4-aligned weight rows

  // Per-thread staging roles
  const int am  = tid >> 1;             // A: row m (0..63)
  const int aks = (tid & 1) * 32;       // A: k-slice base (0 or 32)
  const int bn  = tid & 15;             // B: column n (0..15)
  const int bks = (tid >> 4) * 8;       // B: k-slice base (0..56 step 8)

  // Decode this thread's B output pixel once
  const int bp  = n0 + bn;
  const int bb  = bp / HW;
  const int bq  = bp - bb * HW;
  const int boy = bq / Wout;
  const int box = bq - boy * Wout;
  const int oy0 = boy * stride - pad;   // input origin of this output pixel
  const int ox0 = box * stride - pad;

  v8f acc = {0.f, 0.f, 0.f, 0.f, 0.f, 0.f, 0.f, 0.f};

  for (int k0 = 0; k0 < K; k0 += KC) {
    const bool kfull = (k0 + KC <= K);
    // ---- stage A tile (64xKC weights, fp32 -> bf16) ----
    {
      const float* wrow = w + (size_t)(m0 + am) * K;
      if (kfull) __builtin_prefetch(wrow + k0 + aks + KC, 0, 1);
      if (kfull && krow_aligned) {           // Cout is always a mult of TM here
        const float4* wv = (const float4*)(wrow + k0 + aks);
#pragma unroll
        for (int c = 0; c < 4; ++c)
          *(v8bf*)&As[am][aks + c * 8] = pack8(wv[2 * c], wv[2 * c + 1]);
      } else {
        // clamp-and-select tail: loads are unconditional (batched), result
        // zeroed by cndmask when k is out of range
#pragma unroll
        for (int c = 0; c < 4; ++c) {
          v8bf t;
#pragma unroll
          for (int i = 0; i < 8; ++i) {
            const int k  = k0 + aks + c * 8 + i;
            const int kc = iclamp(k, 0, K - 1);
            float v = wrow[kc];
            t[i] = (bf16_t)((k < K) ? v : 0.f);
          }
          *(v8bf*)&As[am][aks + c * 8] = t;
        }
      }
    }
    // ---- stage B tile (im2col gather, clamp-and-select), n-major ----
    {
      v8bf t;
#pragma unroll
      for (int i = 0; i < 8; ++i) {
        const int k   = k0 + bks + i;
        const int kc  = iclamp(k, 0, K - 1);
        const int cin = kc / khw;
        const int rr  = kc - cin * khw;
        const int kh  = rr / KW;
        const int kw  = rr - kh * KW;
        const int iy  = oy0 + kh;
        const int ix  = ox0 + kw;
        const bool ok = (k < K) & ((unsigned)iy < (unsigned)Hin) &
                        ((unsigned)ix < (unsigned)Win);
        const int iyc = iclamp(iy, 0, Hin - 1);
        const int ixc = iclamp(ix, 0, Win - 1);
        float v = (float)in[(((size_t)bb * Cin + cin) * Hin + iyc) * Win + ixc];
        t[i] = (bf16_t)(ok ? v : 0.f);
      }
      *(v8bf*)&Bs[bn][bks] = t;
    }
    __syncthreads();

    // ---- fragments per documented wave32 bf16 layouts, 2 WMMAs ----
    const int half = lane >> 4;
    const int lm   = lane & 15;
    const int mrow = wave * 16 + lm;
#pragma unroll
    for (int sub = 0; sub < 2; ++sub) {
      v8bf alo = *(const v8bf*)&As[mrow][sub * 32 + half * 8];
      v8bf ahi = *(const v8bf*)&As[mrow][sub * 32 + 16 + half * 8];
      v8bf blo = *(const v8bf*)&Bs[lm][sub * 32 + half * 8];
      v8bf bhi = *(const v8bf*)&Bs[lm][sub * 32 + 16 + half * 8];
      v16bf a  = MERGE16(alo, ahi);
      v16bf bf = MERGE16(blo, bhi);
      acc = __builtin_amdgcn_wmma_f32_16x16x32_bf16(
          false, a, false, bf, (short)0, acc, false, false);
    }
    __syncthreads();
  }

  // ---- epilogue: BN + residual + ReLU, per C/D layout ----
  const int half = lane >> 4;
  const int lm   = lane & 15;
  const int p  = n0 + lm;
  const int b  = p / HW;
  const int q  = p - b * HW;
  const int oy = q / Wout;
  const int ox = q - oy * Wout;
  const bool has_res = (res != nullptr);
#pragma unroll
  for (int r = 0; r < 8; ++r) {
    const int cout = m0 + wave * 16 + half * 8 + r;
    const float inv = bng[cout] / sqrtf(bnv[cout] + 1e-5f);
    float val = acc[r] * inv + (bnb[cout] - bnm[cout] * inv);
    const size_t oidx = (((size_t)b * Cout + cout) * Hout + oy) * Wout + ox;
    if (has_res) val += (float)res[oidx];
    if (relu) val = fmaxf(val, 0.f);
    out[oidx] = (bf16_t)val;
  }
}

// ---------------------------------------------------------------------------
// 3x3 stride-2 pad-1 maxpool (bf16)
// ---------------------------------------------------------------------------
__global__ void maxpool3_kernel(const bf16_t* __restrict__ in,
                                bf16_t* __restrict__ out,
                                int C, int Hin, int Win, int Hout, int Wout,
                                int total) {
  int idx = blockIdx.x * blockDim.x + threadIdx.x;
  if (idx >= total) return;
  int ox = idx % Wout;
  int t  = idx / Wout;
  int oy = t % Hout; t /= Hout;
  int c  = t % C;
  int b  = t / C;
  float m = -3.0e38f;
  for (int dy = 0; dy < 3; ++dy)
    for (int dx = 0; dx < 3; ++dx) {
      int iy = oy * 2 - 1 + dy;
      int ix = ox * 2 - 1 + dx;
      bool ok = (unsigned)iy < (unsigned)Hin && (unsigned)ix < (unsigned)Win;
      int iyc = iclamp(iy, 0, Hin - 1);
      int ixc = iclamp(ix, 0, Win - 1);
      float v = (float)in[(((size_t)b * C + c) * Hin + iyc) * Win + ixc];
      m = fmaxf(m, ok ? v : -3.0e38f);
    }
  out[idx] = (bf16_t)m;
}

// ---------------------------------------------------------------------------
// ROI crop + 2-level SPP (both 2x2) -> (B,R,4096) bf16, channel-major flatten
// ---------------------------------------------------------------------------
__global__ void roi_spp_kernel(const bf16_t* __restrict__ feats,
                               const int* __restrict__ ssw,
                               bf16_t* __restrict__ y,
                               int C, int Hf, int Wf, int Rn, int total) {
  int idx = blockIdx.x * blockDim.x + threadIdx.x;  // over B*R*(C*4)
  if (idx >= total) return;
  const int e = idx & 2047;           // C*2*2 = 2048
  int t = idx >> 11;
  const int r = t % Rn;
  const int b = t / Rn;
  const int c = e >> 2;
  const int i = (e >> 1) & 1;
  const int j = e & 1;
  const int* box = ssw + ((size_t)b * Rn + r) * 4;
  const int r0 = box[0];
  const int c0 = box[1];
  float m = -3.0e38f;
  for (int di = 0; di < 2; ++di)
    for (int dj = 0; dj < 2; ++dj) {
      int yy = r0 + i * 2 + di;
      int xx = c0 + j * 2 + dj;
      m = fmaxf(m, (float)feats[(((size_t)b * C + c) * Hf + yy) * Wf + xx]);
    }
  bf16_t mv = (bf16_t)m;
  const size_t base = ((size_t)b * Rn + r) * 4096;
  y[base + e]        = mv;   // pyramid level 0
  y[base + 2048 + e] = mv;   // pyramid level 1 (identical)
}

// ---------------------------------------------------------------------------
// FC GEMM: out[n][o] = act[n][:] . W[o][:] + bias[o], WMMA bf16, fused ReLU.
// K is always a multiple of KC and rows are 16B aligned (K=4096).
// ---------------------------------------------------------------------------
__global__ __launch_bounds__(128) void gemm_bias_wmma(
    const bf16_t* __restrict__ act, const float* __restrict__ w,
    const float* __restrict__ bias,
    bf16_t* __restrict__ out_bf, float* __restrict__ out_f32,
    int relu, int N, int K, int O) {
  __shared__ bf16_t As[TM][RS];    // [o][k]
  __shared__ bf16_t Bs[TN][RS];    // [n][k]

  const int tid  = threadIdx.x;
  const int wave = tid >> 5;
  const int lane = tid & 31;
  const int n0 = blockIdx.x * TN;
  const int o0 = blockIdx.y * TM;

  const int am  = tid >> 1;
  const int aks = (tid & 1) * 32;
  const int bn  = tid & 15;
  const int bks = (tid >> 4) * 8;
  const bool mfull = (o0 + TM <= O);
  const bool rowok = (o0 + am) < O;
  const int  rowc  = iclamp(o0 + am, 0, O - 1);

  v8f acc = {0.f, 0.f, 0.f, 0.f, 0.f, 0.f, 0.f, 0.f};

  for (int k0 = 0; k0 < K; k0 += KC) {
    // ---- stage A tile (weights fp32 -> bf16) ----
    if (mfull) {
      const float* wrow = w + (size_t)(o0 + am) * K + k0 + aks;
      __builtin_prefetch(wrow + KC, 0, 1);
      const float4* wv = (const float4*)wrow;
#pragma unroll
      for (int c = 0; c < 4; ++c)
        *(v8bf*)&As[am][aks + c * 8] = pack8(wv[2 * c], wv[2 * c + 1]);
    } else {
      // clamp-and-select: unconditional loads from a valid row, zeroed after
      const float* wrow = w + (size_t)rowc * K + k0 + aks;
#pragma unroll
      for (int c = 0; c < 4; ++c) {
        v8bf t;
#pragma unroll
        for (int i = 0; i < 8; ++i) {
          float v = wrow[c * 8 + i];
          t[i] = (bf16_t)(rowok ? v : 0.f);
        }
        *(v8bf*)&As[am][aks + c * 8] = t;
      }
    }
    // ---- stage B tile (dense bf16, vector copy) ----
    {
      v8bf t = *(const v8bf*)(act + (size_t)(n0 + bn) * K + k0 + bks);
      *(v8bf*)&Bs[bn][bks] = t;
    }
    __syncthreads();

    const int half = lane >> 4;
    const int lm   = lane & 15;
    const int mrow = wave * 16 + lm;
#pragma unroll
    for (int sub = 0; sub < 2; ++sub) {
      v8bf alo = *(const v8bf*)&As[mrow][sub * 32 + half * 8];
      v8bf ahi = *(const v8bf*)&As[mrow][sub * 32 + 16 + half * 8];
      v8bf blo = *(const v8bf*)&Bs[lm][sub * 32 + half * 8];
      v8bf bhi = *(const v8bf*)&Bs[lm][sub * 32 + 16 + half * 8];
      v16bf a  = MERGE16(alo, ahi);
      v16bf bf = MERGE16(blo, bhi);
      acc = __builtin_amdgcn_wmma_f32_16x16x32_bf16(
          false, a, false, bf, (short)0, acc, false, false);
    }
    __syncthreads();
  }

  const int half = lane >> 4;
  const int lm   = lane & 15;
  const int n = n0 + lm;
#pragma unroll
  for (int r = 0; r < 8; ++r) {
    const int o = o0 + wave * 16 + half * 8 + r;
    if (o >= O || n >= N) continue;
    float val = acc[r] + bias[o];
    if (relu) val = fmaxf(val, 0.f);
    const size_t oidx = (size_t)n * O + o;
    if (out_bf)  out_bf[oidx]  = (bf16_t)val;
    if (out_f32) out_f32[oidx] = val;
  }
}

// ---------------------------------------------------------------------------
// Dual-softmax head: sc over classes (2), sd over proposals (64);
// out = [ sum_r sc*sd (8,2) | sd (8,64,2) | sc (8,64,2) ]
// ---------------------------------------------------------------------------
__global__ void wsddn_head_kernel(const float* __restrict__ xc,
                                  const float* __restrict__ xd,
                                  float* __restrict__ out, int Rn) {
  const int b = blockIdx.x;
  const int t = threadIdx.x;
  __shared__ float sc[64][2];
  __shared__ float sdv[64][2];
  if (t < Rn) {
    float a0 = xc[((size_t)b * Rn + t) * 2 + 0];
    float a1 = xc[((size_t)b * Rn + t) * 2 + 1];
    float mx = fmaxf(a0, a1);
    float e0 = expf(a0 - mx), e1 = expf(a1 - mx);
    float s = e0 + e1;
    sc[t][0] = e0 / s;
    sc[t][1] = e1 / s;
    sdv[t][0] = xd[((size_t)b * Rn + t) * 2 + 0];
    sdv[t][1] = xd[((size_t)b * Rn + t) * 2 + 1];
  }
  __syncthreads();
  if (t < Rn) {
    out[16 + 2 * Rn * 8 + ((size_t)b * Rn + t) * 2 + 0] = sc[t][0];
    out[16 + 2 * Rn * 8 + ((size_t)b * Rn + t) * 2 + 1] = sc[t][1];
  }
  if (t < 2) {
    float mx = -3.0e38f;
    for (int r = 0; r < Rn; ++r) mx = fmaxf(mx, sdv[r][t]);
    float s = 0.f;
    for (int r = 0; r < Rn; ++r) s += expf(sdv[r][t] - mx);
    float accv = 0.f;
    for (int r = 0; r < Rn; ++r) {
      float ev = expf(sdv[r][t] - mx) / s;
      out[16 + ((size_t)b * Rn + r) * 2 + t] = ev;  // sd
      accv += sc[r][t] * ev;
    }
    out[(size_t)b * 2 + t] = accv;                  // (sc*sd).sum(axis=1)
  }
}

// ---------------------------------------------------------------------------
// Host orchestration
// ---------------------------------------------------------------------------
namespace {
struct BNP { const float *g, *b, *m, *v; };
struct Blk {
  BNP bn1, bn2, bnd;
  const float *w1, *w2, *wd;
  bool down;
};
}  // namespace

extern "C" void kernel_launch(void* const* d_in, const int* in_sizes, int n_in,
                              void* d_out, int out_size, void* d_ws, size_t ws_size,
                              hipStream_t stream) {
  (void)in_sizes; (void)n_in; (void)out_size; (void)ws_size;
  const int Bn = 8, Rn = 64, IMG = 256;

  // ---- walk d_in in JAX pytree order (dicts sorted by key) ----
  int idx = 0;
  auto nxt = [&]() -> const float* { return (const float*)d_in[idx++]; };
  const float* x_f32 = nxt();                     // 'x'
  const int*   ssw   = (const int*)d_in[idx++];   // 'ssw_get'
  // params: bn1, conv1, fc6, fc7, fc8c, fc8d, layers
  BNP bn1 = {nxt(), nxt(), nxt(), nxt()};
  const float* wconv1 = nxt();
  const float *fc6w = nxt(), *fc6b = nxt();
  const float *fc7w = nxt(), *fc7b = nxt();
  const float *fc8cw = nxt(), *fc8cb = nxt();
  const float *fc8dw = nxt(), *fc8db = nxt();
  const int NBLK[4] = {3, 4, 6, 3};
  const int CHN[4]  = {64, 128, 256, 512};
  Blk blocks[16];
  int nb = 0;
  for (int li = 0; li < 4; ++li)
    for (int bi = 0; bi < NBLK[li]; ++bi) {
      Blk bk{};
      bk.bn1 = {nxt(), nxt(), nxt(), nxt()};
      bk.bn2 = {nxt(), nxt(), nxt(), nxt()};
      bk.w1 = nxt();
      bk.w2 = nxt();
      bk.down = (li > 0 && bi == 0);
      if (bk.down) {
        bk.wd = nxt();
        bk.bnd = {nxt(), nxt(), nxt(), nxt()};
      }
      blocks[nb++] = bk;
    }

  // ---- workspace layout (bf16 activations) ----
  char* ws = (char*)d_ws;
  const size_t BIG = 17u * 1024u * 1024u;  // > 8*64*128*128*2 bytes
  bf16_t* bufA = (bf16_t*)(ws + 0 * BIG);
  bf16_t* bufB = (bf16_t*)(ws + 1 * BIG);
  bf16_t* bufC = (bf16_t*)(ws + 2 * BIG);
  bf16_t* xbf  = (bf16_t*)(ws + 3 * BIG);
  size_t off = 4 * BIG;
  bf16_t* ybuf = (bf16_t*)(ws + off); off += (size_t)Bn * Rn * 4096 * 2;
  bf16_t* h6   = (bf16_t*)(ws + off); off += (size_t)Bn * Rn * 4096 * 2;
  bf16_t* h7   = (bf16_t*)(ws + off); off += (size_t)Bn * Rn * 4096 * 2;
  float*  xc   = (float*)(ws + off);  off += (size_t)Bn * Rn * 2 * 4;
  float*  xd   = (float*)(ws + off);

  // ---- convert input to bf16 ----
  {
    int nx = Bn * 3 * IMG * IMG;
    cvt_f32_bf16<<<(nx + 255) / 256, 256, 0, stream>>>(x_f32, xbf, nx);
  }

  auto conv = [&](const bf16_t* in, bf16_t* out, const float* w, BNP bn,
                  const bf16_t* res, int relu, int Cin, int Cout,
                  int Hin, int Win, int ks, int stride, int pad) {
    int Hout = (Hin + 2 * pad - ks) / stride + 1;
    int Wout = (Win + 2 * pad - ks) / stride + 1;
    dim3 grid((Bn * Hout * Wout) / TN, (Cout + TM - 1) / TM);
    conv_bn_wmma<<<grid, 128, 0, stream>>>(
        in, out, w, bn.g, bn.b, bn.m, bn.v, res, relu,
        Cin, Cout, Hin, Win, Hout, Wout, ks, ks, stride, pad);
  };

  // conv1 (7x7 s2 p3) + BN + ReLU : xbf -> bufB (8,64,128,128)
  conv(xbf, bufB, wconv1, bn1, nullptr, 1, 3, 64, IMG, IMG, 7, 2, 3);
  // maxpool 3x3 s2 p1 : bufB -> bufA (8,64,64,64)
  {
    int total = Bn * 64 * 64 * 64;
    maxpool3_kernel<<<(total + 255) / 256, 256, 0, stream>>>(
        bufB, bufA, 64, 128, 128, 64, 64, total);
  }

  // ---- residual stages: x lives in bufA ----
  int H = 64, Cin = 64;
  nb = 0;
  for (int li = 0; li < 4; ++li) {
    int C = CHN[li];
    for (int bi = 0; bi < NBLK[li]; ++bi) {
      Blk& bk = blocks[nb++];
      int stride = (li > 0 && bi == 0) ? 2 : 1;
      int Hout = (stride == 2) ? H / 2 : H;
      const bf16_t* resptr;
      if (bk.down) {
        // identity = BN(1x1 s2 conv(x)) -> bufC
        conv(bufA, bufC, bk.wd, bk.bnd, nullptr, 0, Cin, C, H, H, 1, stride, 0);
        resptr = bufC;
      } else {
        resptr = bufA;  // in-place residual (1:1 element ownership)
      }
      // out1 = ReLU(BN(conv3x3 s)) : bufA -> bufB
      conv(bufA, bufB, bk.w1, bk.bn1, nullptr, 1, Cin, C, H, H, 3, stride, 1);
      // out  = ReLU(BN(conv3x3 1) + identity) : bufB -> bufA
      conv(bufB, bufA, bk.w2, bk.bn2, resptr, 1, C, C, Hout, Hout, 3, 1, 1);
      H = Hout;
      Cin = C;
    }
  }
  // feats = bufA : (8,512,8,8)

  // ---- ROI + SPP -> y (8,64,4096) bf16 ----
  {
    int total = Bn * Rn * 2048;
    roi_spp_kernel<<<(total + 255) / 256, 256, 0, stream>>>(
        bufA, ssw, ybuf, 512, 8, 8, Rn, total);
  }

  // ---- FC stack (WMMA GEMMs) ----
  const int Nrow = Bn * Rn;  // 512
  {
    dim3 g6(Nrow / TN, 4096 / TM);
    gemm_bias_wmma<<<g6, 128, 0, stream>>>(ybuf, fc6w, fc6b, h6, nullptr, 1,
                                           Nrow, 4096, 4096);
    gemm_bias_wmma<<<g6, 128, 0, stream>>>(h6, fc7w, fc7b, h7, nullptr, 1,
                                           Nrow, 4096, 4096);
  }
  {
    dim3 g8(Nrow / TN, 1);
    gemm_bias_wmma<<<g8, 128, 0, stream>>>(h7, fc8cw, fc8cb, nullptr, xc, 1,
                                           Nrow, 4096, 2);
    gemm_bias_wmma<<<g8, 128, 0, stream>>>(h7, fc8dw, fc8db, nullptr, xd, 1,
                                           Nrow, 4096, 2);
  }

  // ---- dual softmax + combine -> d_out (16 + 1024 + 1024 fp32) ----
  wsddn_head_kernel<<<Bn, 64, 0, stream>>>(xc, xd, (float*)d_out, Rn);
}
